// SetAbstraction_42683384987906
// MI455X (gfx1250) — compile-verified, matrix operand order checked by reading, not run on hardware
//
#include <hip/hip_runtime.h>
#include <cstdint>
#include <cstddef>

typedef _Float16 half_t;
typedef _Float16 v16h __attribute__((ext_vector_type(16)));
typedef float    v8f  __attribute__((ext_vector_type(8)));

#define OUT_OFF 49152   // 16*1024*3 new_xyz floats precede new_features in d_out

__device__ __forceinline__ v8f wmma16(v16h a, v16h b, v8f c) {
    // D(16x16 f32) = A(16x32 f16) * B(32x16 f16) + C
    return __builtin_amdgcn_wmma_f32_16x16x32_f16(false, a, false, b, (short)0, c, false, false);
}

// A/B fragment: lane<16 holds K {0..7,16..23}+kbase, lane>=16 holds K {8..15,24..31}+kbase
__device__ __forceinline__ v16h load_frag(const half_t* row, int kbase, int hi) {
    v16h f;
    const half_t* p0 = row + kbase + hi * 8;
    const half_t* p1 = row + kbase + 16 + hi * 8;
#pragma unroll
    for (int e = 0; e < 8; ++e) f[e] = p0[e];
#pragma unroll
    for (int e = 0; e < 8; ++e) f[8 + e] = p1[e];
    return f;
}

__device__ __forceinline__ float gelu_exact(float x) {
    return 0.5f * x * (1.0f + erff(x * 0.70710678118654752f));
}

// ---- 16-lane (DPP row) reductions on the VALU pipe; DPP rows == WMMA C half-groups.
#if defined(__has_builtin)
#if __has_builtin(__builtin_amdgcn_mov_dpp)
#define HAVE_DPP 1
#endif
#endif

#ifdef HAVE_DPP
template <int CTRL>
__device__ __forceinline__ float dpp_f32(float x) {
    return __int_as_float(__builtin_amdgcn_mov_dpp(__float_as_int(x), CTRL, 0xF, 0xF, true));
}
__device__ __forceinline__ float sum16(float x) {
    x += dpp_f32<0xB1>(x);   // quad_perm [1,0,3,2]  (xor 1)
    x += dpp_f32<0x4E>(x);   // quad_perm [2,3,0,1]  (xor 2)
    x += dpp_f32<0x124>(x);  // row_ror:4
    x += dpp_f32<0x128>(x);  // row_ror:8
    return x;                // all 16 lanes of the row hold the total
}
__device__ __forceinline__ float max16(float x) {
    x = fmaxf(x, dpp_f32<0xB1>(x));
    x = fmaxf(x, dpp_f32<0x4E>(x));
    x = fmaxf(x, dpp_f32<0x124>(x));
    x = fmaxf(x, dpp_f32<0x128>(x));
    return x;
}
#else
__device__ __forceinline__ float sum16(float x) {
    for (int m = 8; m >= 1; m >>= 1) x += __shfl_xor(x, m);
    return x;
}
__device__ __forceinline__ float max16(float x) {
    for (int m = 8; m >= 1; m >>= 1) x = fmaxf(x, __shfl_xor(x, m));
    return x;
}
#endif

// ---------------------------------------------------------------------------
// Farthest point sampling: one block per batch, xyz + dist resident in LDS.
// ---------------------------------------------------------------------------
__global__ __launch_bounds__(256) void fps_kernel(const float* __restrict__ xyz,
                                                  int* __restrict__ fps_idx,
                                                  float* __restrict__ newxyz,
                                                  float* __restrict__ dout) {
    __shared__ float xs[4096], ys[4096], zs[4096], dist[4096];
    __shared__ float rv[8];
    __shared__ int   ri[8];
    __shared__ int   curs;
    const int b = blockIdx.x, tid = threadIdx.x;
    for (int p = tid; p < 4096; p += 256) {
        const float* q = xyz + ((size_t)b * 4096 + p) * 3;
        xs[p] = q[0]; ys[p] = q[1]; zs[p] = q[2];
        dist[p] = 1e10f;
    }
    __syncthreads();
    int cur = 0;
    for (int i = 0; i < 1024; ++i) {
        if (tid == 0) {
            fps_idx[b * 1024 + i] = cur;
            float* o  = newxyz + (size_t)(b * 1024 + i) * 3;
            float* o2 = dout   + (size_t)(b * 1024 + i) * 3;
            o[0] = o2[0] = xs[cur]; o[1] = o2[1] = ys[cur]; o[2] = o2[2] = zs[cur];
        }
        const float cx = xs[cur], cy = ys[cur], cz = zs[cur];
        float bv = -1.0f; int bi = 0;
        for (int p = tid; p < 4096; p += 256) {
            const float dx = xs[p] - cx, dy = ys[p] - cy, dz = zs[p] - cz;
            const float d  = dx * dx + dy * dy + dz * dz;
            const float nd = fminf(dist[p], d);
            dist[p] = nd;
            if (nd > bv) { bv = nd; bi = p; }
        }
        for (int m = 16; m >= 1; m >>= 1) {   // wave32 argmax, lowest index on ties
            const float ov = __shfl_xor(bv, m);
            const int   oi = __shfl_xor(bi, m);
            if (ov > bv || (ov == bv && oi < bi)) { bv = ov; bi = oi; }
        }
        if ((tid & 31) == 0) { rv[tid >> 5] = bv; ri[tid >> 5] = bi; }
        __syncthreads();
        if (tid == 0) {
            float BV = rv[0]; int BI = ri[0];
            for (int w = 1; w < 8; ++w)
                if (rv[w] > BV || (rv[w] == BV && ri[w] < BI)) { BV = rv[w]; BI = ri[w]; }
            curs = BI;
        }
        __syncthreads();
        cur = curs;
    }
}

// ---------------------------------------------------------------------------
// Ball query: one wave per centroid; first-k in ascending index, pad w/ first.
// ---------------------------------------------------------------------------
template <int KNB>
__global__ __launch_bounds__(256) void ballquery_kernel(const float* __restrict__ xyz,
                                                        const float* __restrict__ newxyz,
                                                        int* __restrict__ nbr, float r2) {
    const int wid  = (int)((blockIdx.x * blockDim.x + threadIdx.x) >> 5);
    const int lane = threadIdx.x & 31;
    if (wid >= 16384) return;
    const int b = wid >> 10;
    const float cx = newxyz[wid * 3 + 0], cy = newxyz[wid * 3 + 1], cz = newxyz[wid * 3 + 2];
    const float* base = xyz + (size_t)b * 4096 * 3;
    int count = 0, first = -1;
    for (int p0 = 0; p0 < 4096 && count < KNB; p0 += 32) {
        const int p = p0 + lane;
        const float dx = base[p * 3 + 0] - cx, dy = base[p * 3 + 1] - cy, dz = base[p * 3 + 2] - cz;
        const bool pred = (dx * dx + dy * dy + dz * dz) <= r2;
        const unsigned mlo = (unsigned)__ballot(pred);
        const int pos = count + __popc(mlo & ((1u << lane) - 1u));
        if (pred && pos < KNB) nbr[(size_t)wid * KNB + pos] = p;
        if (first < 0 && mlo) first = p0 + __ffs(mlo) - 1;
        count += __popc(mlo);
    }
    if (count < KNB)
        for (int t = count + lane; t < KNB; t += 32) nbr[(size_t)wid * KNB + t] = first;
}

// ---------------------------------------------------------------------------
// Fused gather + MLP kernel, 3 recompute phases (global norm needs 2 stat passes).
//   PHASE 0: GEMM1 -> per-channel sum/sumsq (register partials)
//   PHASE 1: GEMM1 -> norm1+gelu -> GEMM2 -> sum/sumsq (register partials)
//   PHASE 2: full pipeline -> norm2+gelu -> max over k -> d_out
// 128 threads = 4 waves; each wave owns one 16-column WMMA tile (64 cols/chunk,
// chunks are whole centroids so the k-max is block-local). Stats accumulate in
// per-lane registers across all chunks; the 16-lane fold happens ONCE at kernel
// end via DPP, keeping the hot loop free of cross-lane ops.
// ---------------------------------------------------------------------------
template <int C1, int C2, int KNB, int PHASE>
__global__ __launch_bounds__(128) void mlp_kernel(const float* __restrict__ xyz,
                                                  const float* __restrict__ feats,
                                                  const float* __restrict__ w1,
                                                  const float* __restrict__ w2,
                                                  const int* __restrict__ nbr,
                                                  const float* __restrict__ newxyz,
                                                  const float* __restrict__ fin1,
                                                  const float* __restrict__ fin2,
                                                  float* __restrict__ partials,
                                                  float* __restrict__ dout, int chBase) {
    constexpr int NCH = (16384 * KNB) / 64;  // column chunks of 64
    constexpr int CS  = (PHASE == 0) ? C1 : C2;  // stat channel count
    __shared__ half_t Xt[64][104];                       // gathered input, col-contiguous
    __shared__ half_t Wa[C1][96];                        // layer-1 weights, K padded 67->96
    __shared__ half_t Wb[(PHASE >= 1) ? C2 : 1][C1];     // layer-2 weights
    __shared__ half_t Y1t[(PHASE >= 1) ? 64 : 1][C1 + 8];
    __shared__ float  lsW[(PHASE < 2) ? 4 : 1][2 * CS];  // per-wave folded stats
    __shared__ float  f1buf[2 * C1];
    __shared__ float  f2buf[2 * C2];
    __shared__ float  pmax[(PHASE == 2) ? 4 : 1][C2];

    const int tid = threadIdx.x;
    const int lane = tid & 31, wid = tid >> 5;
    const int sub = lane & 15, hi = lane >> 4;

    for (int t = tid; t < C1 * 96; t += 128) {
        const int r = t / 96, c = t % 96;
        Wa[r][c] = (c < 67) ? (half_t)w1[r * 67 + c] : (half_t)0.0f;
    }
    if constexpr (PHASE >= 1) {
        for (int t = tid; t < C2 * C1; t += 128) Wb[t / C1][t % C1] = (half_t)w2[t];
        for (int t = tid; t < 2 * C1; t += 128) f1buf[t] = fin1[t];
    }
    if constexpr (PHASE == 2)
        for (int t = tid; t < 2 * C2; t += 128) f2buf[t] = fin2[t];

    // per-lane stat accumulators (stay in VGPRs for the whole kernel)
    float accS[(PHASE < 2) ? CS / 16 : 1][8];
    float accQ[(PHASE < 2) ? CS / 16 : 1][8];
    if constexpr (PHASE < 2) {
#pragma unroll
        for (int t = 0; t < CS / 16; ++t)
#pragma unroll
            for (int r = 0; r < 8; ++r) { accS[t][r] = 0.0f; accQ[t][r] = 0.0f; }
    }

    for (int chunk = blockIdx.x; chunk < NCH; chunk += gridDim.x) {
        __syncthreads();
        {   // gather 64 columns of [xyz_rel(3) | feat(64) | pad->96] as f16
            const int j = tid >> 1, part = tid & 1;
            const int colg = chunk * 64 + j;
            const int cent = colg / KNB;
            const int bb = cent >> 10;
            const int idx = nbr[colg];
            const float4* fp4 = (const float4*)(feats + (((size_t)bb << 12) + idx) * 64);
            if (part == 0) {
                const float* xp = xyz + (((size_t)bb << 12) + idx) * 3;
                const float* cp = newxyz + (size_t)cent * 3;
                Xt[j][0] = (half_t)(xp[0] - cp[0]);
                Xt[j][1] = (half_t)(xp[1] - cp[1]);
                Xt[j][2] = (half_t)(xp[2] - cp[2]);
#pragma unroll
                for (int q = 0; q < 8; ++q) {     // feat[0..31] -> rows 3..34 (b128 loads)
                    const float4 v = fp4[q];
                    Xt[j][3 + 4 * q + 0] = (half_t)v.x;
                    Xt[j][3 + 4 * q + 1] = (half_t)v.y;
                    Xt[j][3 + 4 * q + 2] = (half_t)v.z;
                    Xt[j][3 + 4 * q + 3] = (half_t)v.w;
                }
                __builtin_prefetch(nbr + (size_t)(chunk + gridDim.x) * 64 + j, 0, 3);
            } else {
#pragma unroll
                for (int q = 8; q < 16; ++q) {    // feat[32..63] -> rows 35..66
                    const float4 v = fp4[q];
                    Xt[j][3 + 4 * q + 0] = (half_t)v.x;
                    Xt[j][3 + 4 * q + 1] = (half_t)v.y;
                    Xt[j][3 + 4 * q + 2] = (half_t)v.z;
                    Xt[j][3 + 4 * q + 3] = (half_t)v.w;
                }
#pragma unroll
                for (int r = 67; r < 96; ++r) Xt[j][r] = (half_t)0.0f;  // K padding
            }
        }
        __syncthreads();

        // ---- GEMM1: y1 = W1 (C1x67) @ X (67x64cols) ----
        v8f y1[C1 / 16];
#pragma unroll
        for (int rt = 0; rt < C1 / 16; ++rt) {
            v8f acc = {0.f, 0.f, 0.f, 0.f, 0.f, 0.f, 0.f, 0.f};
#pragma unroll
            for (int kb = 0; kb < 3; ++kb) {
                const v16h a = load_frag(&Wa[rt * 16 + sub][0], kb * 32, hi);
                const v16h bfr = load_frag(&Xt[wid * 16 + sub][0], kb * 32, hi);
                acc = wmma16(a, bfr, acc);
            }
            y1[rt] = acc;
        }

        if constexpr (PHASE == 0) {
#pragma unroll
            for (int rt = 0; rt < C1 / 16; ++rt)
#pragma unroll
                for (int r = 0; r < 8; ++r) {
                    const float v = y1[rt][r];
                    accS[rt][r] += v;
                    accQ[rt][r] = fmaf(v, v, accQ[rt][r]);
                }
        } else {
            // norm1 + gelu, restage as f16 B-operand for GEMM2 (wave-private cols)
#pragma unroll
            for (int rt = 0; rt < C1 / 16; ++rt)
#pragma unroll
                for (int r = 0; r < 8; ++r) {
                    const int ch = rt * 16 + hi * 8 + r;
                    float v = y1[rt][r] * f1buf[ch] + f1buf[C1 + ch];
                    v = gelu_exact(v);
                    Y1t[wid * 16 + sub][ch] = (half_t)v;
                }
            // ---- GEMM2: y2 = W2 (C2xC1) @ y1n (C1x64cols) ----
#pragma unroll
            for (int rt2 = 0; rt2 < C2 / 16; ++rt2) {
                v8f acc = {0.f, 0.f, 0.f, 0.f, 0.f, 0.f, 0.f, 0.f};
#pragma unroll
                for (int kb = 0; kb < C1 / 32; ++kb) {
                    const v16h a = load_frag(&Wb[rt2 * 16 + sub][0], kb * 32, hi);
                    const v16h bfr = load_frag(&Y1t[wid * 16 + sub][0], kb * 32, hi);
                    acc = wmma16(a, bfr, acc);
                }
                if constexpr (PHASE == 1) {
#pragma unroll
                    for (int r = 0; r < 8; ++r) {
                        const float v = acc[r];
                        accS[rt2][r] += v;
                        accQ[rt2][r] = fmaf(v, v, accQ[rt2][r]);
                    }
                } else {
#pragma unroll
                    for (int r = 0; r < 8; ++r) {
                        const int ch = rt2 * 16 + hi * 8 + r;
                        float v = acc[r] * f2buf[ch] + f2buf[C2 + ch];
                        v = max16(gelu_exact(v));         // max over this col-tile's 16 kk
                        if (sub == 0) pmax[wid][ch] = v;
                    }
                }
            }
            if constexpr (PHASE == 2) {
                __syncthreads();
                constexpr int CPC = 64 / KNB;   // whole centroids per chunk
                constexpr int TPC = KNB / 16;   // col-tiles per centroid
                for (int t = tid; t < CPC * C2; t += 128) {
                    const int cl = t / C2, ch = t % C2;
                    float mv = pmax[cl * TPC][ch];
#pragma unroll
                    for (int q = 1; q < TPC; ++q) mv = fmaxf(mv, pmax[cl * TPC + q][ch]);
                    const int cg = chunk * CPC + cl;
                    const int bb = cg >> 10, mm = cg & 1023;
                    dout[OUT_OFF + (((size_t)bb * 320 + chBase + ch) << 10) + mm] = mv;
                }
            }
        }
    }

    if constexpr (PHASE < 2) {   // one-time 16-lane fold (DPP) + deterministic block reduce
        __syncthreads();
#pragma unroll
        for (int rt = 0; rt < CS / 16; ++rt)
#pragma unroll
            for (int r = 0; r < 8; ++r) {
                const float s = sum16(accS[rt][r]);
                const float q = sum16(accQ[rt][r]);
                if (sub == 0) {
                    const int ch = rt * 16 + hi * 8 + r;
                    lsW[wid][ch] = s;
                    lsW[wid][CS + ch] = q;
                }
            }
        __syncthreads();
        for (int t = tid; t < 2 * CS; t += 128) {
            float s = 0.f;
#pragma unroll
            for (int w = 0; w < 4; ++w) s += lsW[w][t];
            partials[(size_t)blockIdx.x * 2 * CS + t] = s;
        }
    }
}

// Fixed-order block-partial reduction + fold norm into per-channel scale/shift.
__global__ void reduce_finalize(const float* __restrict__ partials, int nblk, int C,
                                float cnt, const float* __restrict__ g,
                                const float* __restrict__ b, float* __restrict__ fin) {
    const int t = threadIdx.x;
    if (t >= C) return;
    float s = 0.f, q = 0.f;
    for (int i = 0; i < nblk; ++i) {
        s += partials[(size_t)i * 2 * C + t];
        q += partials[(size_t)i * 2 * C + C + t];
    }
    const float mean = s / cnt;
    const float var  = q / cnt - mean * mean;
    const float rstd = rsqrtf(var + 1e-5f);
    const float sc = g[t] * rstd;
    fin[t] = sc;
    fin[C + t] = b[t] - mean * sc;
}

template <int C1, int C2, int KNB>
static void run_scale(const float* xyz, const float* feats, const float* w1, const float* w2,
                      const float* g1, const float* b1, const float* g2, const float* b2,
                      const int* nbr, const float* newxyz, float* partials,
                      float* fin1, float* fin2, float* dout, int chBase, hipStream_t stream) {
    constexpr int NBLK = 512;
    const float cnt = 16.0f * 1024.0f * (float)KNB;
    mlp_kernel<C1, C2, KNB, 0><<<NBLK, 128, 0, stream>>>(xyz, feats, w1, w2, nbr, newxyz,
                                                         nullptr, nullptr, partials, dout, chBase);
    reduce_finalize<<<1, C1, 0, stream>>>(partials, NBLK, C1, cnt, g1, b1, fin1);
    mlp_kernel<C1, C2, KNB, 1><<<NBLK, 128, 0, stream>>>(xyz, feats, w1, w2, nbr, newxyz,
                                                         fin1, nullptr, partials, dout, chBase);
    reduce_finalize<<<1, C2, 0, stream>>>(partials, NBLK, C2, cnt, g2, b2, fin2);
    mlp_kernel<C1, C2, KNB, 2><<<NBLK, 128, 0, stream>>>(xyz, feats, w1, w2, nbr, newxyz,
                                                         fin1, fin2, partials, dout, chBase);
}

extern "C" void kernel_launch(void* const* d_in, const int* in_sizes, int n_in,
                              void* d_out, int out_size, void* d_ws, size_t ws_size,
                              hipStream_t stream) {
    (void)in_sizes; (void)n_in; (void)out_size; (void)ws_size;
    const float* xyz   = (const float*)d_in[0];
    const float* feats = (const float*)d_in[1];
    const float* w00 = (const float*)d_in[2];
    const float* w01 = (const float*)d_in[3];
    const float* w10 = (const float*)d_in[4];
    const float* w11 = (const float*)d_in[5];
    const float* w20 = (const float*)d_in[6];
    const float* w21 = (const float*)d_in[7];
    // setup_inputs dict order: g00,b00,g01,b01,g10,b10,g11,b11,g20,b20,g21,b21
    const float* g00 = (const float*)d_in[8];
    const float* b00 = (const float*)d_in[9];
    const float* g01 = (const float*)d_in[10];
    const float* b01 = (const float*)d_in[11];
    const float* g10 = (const float*)d_in[12];
    const float* b10 = (const float*)d_in[13];
    const float* g11 = (const float*)d_in[14];
    const float* b11 = (const float*)d_in[15];
    const float* g20 = (const float*)d_in[16];
    const float* b20 = (const float*)d_in[17];
    const float* g21 = (const float*)d_in[18];
    const float* b21 = (const float*)d_in[19];
    float* out = (float*)d_out;

    char* ws = (char*)d_ws;
    int*   fps_idx  = (int*)ws;   ws += (size_t)16384 * sizeof(int);
    float* newxyz   = (float*)ws; ws += (size_t)16 * 1024 * 3 * sizeof(float);
    int*   nbr0     = (int*)ws;   ws += (size_t)16384 * 16 * sizeof(int);
    int*   nbr1     = (int*)ws;   ws += (size_t)16384 * 32 * sizeof(int);
    int*   nbr2     = (int*)ws;   ws += (size_t)16384 * 64 * sizeof(int);
    float* partials = (float*)ws; ws += (size_t)512 * 2 * 128 * sizeof(float);
    float* fin      = (float*)ws; // 6 * 256 floats

    fps_kernel<<<16, 256, 0, stream>>>(xyz, fps_idx, newxyz, out);
    ballquery_kernel<16><<<2048, 256, 0, stream>>>(xyz, newxyz, nbr0, 0.01f);
    ballquery_kernel<32><<<2048, 256, 0, stream>>>(xyz, newxyz, nbr1, 0.04f);
    ballquery_kernel<64><<<2048, 256, 0, stream>>>(xyz, newxyz, nbr2, 0.16f);

    run_scale<32, 64, 16>(xyz, feats, w00, w01, g00, b00, g01, b01, nbr0, newxyz,
                          partials, fin + 0,    fin + 256,  out, 0,   stream);
    run_scale<64, 128, 32>(xyz, feats, w10, w11, g10, b10, g11, b11, nbr1, newxyz,
                           partials, fin + 512,  fin + 768,  out, 64,  stream);
    run_scale<64, 128, 64>(xyz, feats, w20, w21, g20, b20, g21, b21, nbr2, newxyz,
                           partials, fin + 1024, fin + 1280, out, 192, stream);
}
